// SparseUNet3D_65670049956014
// MI455X (gfx1250) — compile-verified, matrix operand order checked by reading, not run on hardware
//
#include <hip/hip_runtime.h>

// ---------------------------------------------------------------------------
// CDNA5 (gfx1250) types for WMMA
// ---------------------------------------------------------------------------
typedef __attribute__((ext_vector_type(16))) __bf16 v16bf;
typedef __attribute__((ext_vector_type(8)))  float  v8f;

union AFrag { v16bf v; unsigned int w[8]; };
union BF2   { __bf16 b[2]; unsigned int u; };

__device__ inline unsigned int pack_bf(float v0, float v1) {
    BF2 c;
    c.b[0] = (__bf16)v0;      // native cvt (RNE) -> packed bf16
    c.b[1] = (__bf16)v1;
    return c.u;
}

// ---------------------------------------------------------------------------
// Implicit-GEMM conv, tap-major K ordering: K = tap*Cin + ci.
//   kmode 0: 3x3x3, pad 1, stride 1 (27 taps = 9 rows x 3 x-taps)
//   kmode 1: 2x2x2, stride 2        ( 8 taps = 4 rows x 2 x-taps)
// Block = 64 threads = 2 waves; M-block = 32 voxels (x-run), N-block = 32 ch.
// Per step we stage ONE raw input halo row (all x-taps share it) + the B
// tiles of its x-taps; A-fragments for each x-tap are just offset LDS reads
// of the raw row (layout [x][ci-pair] == fragment layout). Each wave then
// issues 6 (3 taps x 2 M-subtiles) v_wmma_f32_16x16x32_bf16 per barrier.
// LDS is double-buffered: stage(step+1) runs before this step's WMMAs.
// Fused BN(scale,shift)+ReLU+mask epilogue; concat via (inA,inB,cinA) split.
// ---------------------------------------------------------------------------
__global__ __launch_bounds__(64)
void conv_wmma_kernel(const float* __restrict__ inA, const float* __restrict__ inB,
                      int cinA, int cinTot, int TAPS,
                      const float* __restrict__ wgt,
                      const float* __restrict__ scale, const float* __restrict__ shift,
                      const float* __restrict__ mask, float* __restrict__ out,
                      int D, int H, int W, int kmode,
                      int ciShift, int ngShift) {
    __shared__ unsigned int lR[2][64][16];     // [buf][x][ci/2] raw halo row
    __shared__ unsigned int lB[2][3][32][16];  // [buf][xtap][n][ci/2]

    // grid: x = xTiles * nGroups (nGroups = 2^ngShift), y = H, z = D
    const int xt = blockIdx.x >> ngShift;
    const int ng = blockIdx.x & ((1 << ngShift) - 1);
    const int yo = blockIdx.y;
    const int zo = blockIdx.z;
    const int x0 = xt << 5;
    const int n0 = ng << 5;

    const int tid  = threadIdx.x;
    const int lane = tid & 31;
    const int wv   = tid >> 5;            // wave -> N subtile (0..1)
    const int half = (lane >> 4) & 1;

    const int Din = (kmode == 1) ? (D << 1) : D;
    const int Hin = (kmode == 1) ? (H << 1) : H;
    const int Win = (kmode == 1) ? (W << 1) : W;
    const int planeIn = Din * Hin * Win;
    const int K = cinTot * TAPS;          // weight row stride (OIDHW flatten)
    const int ciMask = (1 << ciShift) - 1;

    const int ndx   = (kmode == 1) ? 2 : 3;   // x-taps sharing one raw row
    const int xstep = (kmode == 1) ? 2 : 1;
    const int NZY   = (kmode == 1) ? 4 : 9;   // (z,y) tap-row pairs
    const int S     = NZY << ciShift;         // total stage steps

    // staging roles (fixed per thread)
    const int ch = (tid >> 5) << 4;       // first ci of this thread's half

    v8f acc0 = {};
    v8f acc1 = {};

    auto stage = [&](int step, int buf) {
        int zy  = step >> ciShift;
        int ci0 = (step & ciMask) << 5;
        int zz, yy;
        if (kmode == 1) { zz = (zo << 1) + (zy >> 1); yy = (yo << 1) + (zy & 1); }
        else            { zz = zo + zy / 3 - 1;       yy = yo + zy % 3 - 1; }
        bool rowOK = (unsigned)zz < (unsigned)Din && (unsigned)yy < (unsigned)Hin;
        int rowBase = (zz * Hin + yy) * Win;
        // segment select: scalar per step (split aligned to 32)
        const float* srcBase;
        int ciOff;
        if (ci0 < cinA) { srcBase = inA; ciOff = ci0; }
        else            { srcBase = inB; ciOff = ci0 - cinA; }
        int cmax = cinTot - ci0; if (cmax > 32) cmax = 32;
        int gxBase = (kmode == 1) ? (x0 << 1) : (x0 - 1);

        // ---- stage raw halo row: loads x-contiguous across the wave ----
        auto row = [&](int xi) {
            int gx = gxBase + xi;
            bool ok = rowOK && (unsigned)gx < (unsigned)Win;
            const float* p = srcBase + (size_t)(ciOff + ch) * planeIn + rowBase + gx;
            #pragma unroll
            for (int j = 0; j < 8; ++j) {
                int ci = ch + (j << 1);
                float v0 = (ok && (ci     < cmax)) ? p[0]       : 0.0f;
                float v1 = (ok && (ci + 1 < cmax)) ? p[planeIn] : 0.0f;
                p += planeIn * 2;
                lR[buf][xi][(ch >> 1) + j] = pack_bf(v0, v1);
            }
        };
        int xi = tid & 31;
        row(xi);                                   // x 0..31
        if (kmode == 1)      row(xi + 32);         // stride-2 row is 64 wide
        else if (xi < 2)     row(xi + 32);         // halo x 32,33

        // ---- stage B tiles of this row's x-taps: k-index = ci*TAPS + tap ----
        {
            int nl = tid & 31;
            int tapBase = zy * ndx;
            for (int dx = 0; dx < ndx; ++dx) {
                const float* q = wgt + (size_t)(n0 + nl) * K
                                 + (size_t)(ci0 + ch) * TAPS + tapBase + dx;
                #pragma unroll
                for (int j = 0; j < 8; ++j) {
                    int ci = ch + (j << 1);
                    float v0 = (ci     < cmax) ? q[0]    : 0.0f;
                    float v1 = (ci + 1 < cmax) ? q[TAPS] : 0.0f;
                    q += TAPS * 2;
                    lB[buf][dx][nl][(ch >> 1) + j] = pack_bf(v0, v1);
                }
            }
        }
    };

    stage(0, 0);
    __syncthreads();

    const int mb0 = (lane & 15) * xstep;           // raw-row x of M-subtile 0
    const int mb1 = mb0 + (xstep << 4);            // raw-row x of M-subtile 1
    const int nr  = (wv << 4) + (lane & 15);
    const int h4  = half << 2;

    for (int step = 0; step < S; ++step) {
        int buf = step & 1;
        if (step + 1 < S) stage(step + 1, buf ^ 1);   // overlap next loads

        auto fragA = [&](int xoff) {
            AFrag a;
            #pragma unroll
            for (int v = 0; v < 4; ++v) {
                a.w[v]     = lR[buf][xoff][v + h4];
                a.w[4 + v] = lR[buf][xoff][8 + v + h4];
            }
            return a;
        };
        auto doTap = [&](int dx) {
            AFrag a0 = fragA(mb0 + dx);
            AFrag a1 = fragA(mb1 + dx);
            AFrag b;
            #pragma unroll
            for (int v = 0; v < 4; ++v) {
                b.w[v]     = lB[buf][dx][nr][v + h4];
                b.w[4 + v] = lB[buf][dx][nr][8 + v + h4];
            }
            acc0 = __builtin_amdgcn_wmma_f32_16x16x32_bf16(
                       false, a0.v, false, b.v, (short)0, acc0, false, false);
            acc1 = __builtin_amdgcn_wmma_f32_16x16x32_bf16(
                       false, a1.v, false, b.v, (short)0, acc1, false, false);
        };
        doTap(0);
        doTap(1);
        if (ndx == 3) doTap(2);

        __syncthreads();
    }

    // ---- fused BN + ReLU + mask epilogue (C/D layout: lane=N, r=M) ----
    int n = n0 + (wv << 4) + (lane & 15);
    float s  = scale[n];
    float sh = shift[n];
    int plane = D * H * W;
    int rowOut = (zo * H + yo) * W;
    #pragma unroll
    for (int r = 0; r < 8; ++r) {
        int m0v = r + (half << 3);          // msub 0
        int xo0 = x0 + m0v;
        if (xo0 < W) {
            int vox = rowOut + xo0;
            out[n * plane + vox] = fmaxf(acc0[r] * s + sh, 0.0f) * mask[vox];
        }
        int m1v = 16 + m0v;                 // msub 1
        int xo1 = x0 + m1v;
        if (xo1 < W) {
            int vox = rowOut + xo1;
            out[n * plane + vox] = fmaxf(acc1[r] * s + sh, 0.0f) * mask[vox];
        }
    }
}

// ---------------------------------------------------------------------------
// Transpose "up" conv (kernel 2 stride 2): tiny FLOPs -> VALU.
// ---------------------------------------------------------------------------
__global__ __launch_bounds__(256)
void upconv_kernel(const float* __restrict__ in, const float* __restrict__ wgt,
                   const float* __restrict__ mask, float* __restrict__ out,
                   int Cin, int Cout, int D, int H, int W) {
    int plane = D * H * W;
    int idx = blockIdx.x * 256 + threadIdx.x;
    if (idx >= Cout * plane) return;
    int o   = idx / plane;
    int vox = idx - o * plane;
    int z = vox / (H * W);
    int rem = vox - z * H * W;
    int y = rem / W;
    int x = rem - y * W;
    int tap = ((z & 1) << 2) | ((y & 1) << 1) | (x & 1);
    int planeIn = plane >> 3;
    int invox = ((z >> 1) * (H >> 1) + (y >> 1)) * (W >> 1) + (x >> 1);
    float acc = 0.0f;
    for (int ci = 0; ci < Cin; ++ci)
        acc += in[ci * planeIn + invox] * wgt[(o * Cin + ci) * 8 + tap];
    out[idx] = acc * mask[vox];
}

// ---------------------------------------------------------------------------
// 1x1x1 pred head: 32 -> 4 channels, +bias, *mask
// ---------------------------------------------------------------------------
__global__ __launch_bounds__(256)
void pred_kernel(const float* __restrict__ in, const float* __restrict__ w,
                 const float* __restrict__ b, const float* __restrict__ mask,
                 float* __restrict__ out, int plane) {
    int vox = blockIdx.x * 256 + threadIdx.x;
    if (vox >= plane) return;
    float m = mask[vox];
    float xi[32];
    #pragma unroll
    for (int c = 0; c < 32; ++c) xi[c] = in[c * plane + vox];
    #pragma unroll
    for (int o = 0; o < 4; ++o) {
        float acc = b[o];
        #pragma unroll
        for (int c = 0; c < 32; ++c) acc += xi[c] * w[o * 32 + c];
        out[o * plane + vox] = acc * m;
    }
}

// ---------------------------------------------------------------------------
// Mask pyramid
// ---------------------------------------------------------------------------
__global__ __launch_bounds__(256)
void mask_init_kernel(const unsigned char* __restrict__ src,
                      float* __restrict__ dst, int n) {
    int i = blockIdx.x * 256 + threadIdx.x;
    if (i < n) dst[i] = src[i] ? 1.0f : 0.0f;
}

__global__ __launch_bounds__(256)
void mask_pool_kernel(const float* __restrict__ fine, float* __restrict__ coarse,
                      int D, int H, int W) {   // coarse dims; fine is 2x
    int i = blockIdx.x * 256 + threadIdx.x;
    int plane = D * H * W;
    if (i >= plane) return;
    int z = i / (H * W);
    int rem = i - z * H * W;
    int y = rem / W;
    int x = rem - y * W;
    int Hf = H << 1, Wf = W << 1;
    float v = 0.0f;
    #pragma unroll
    for (int a = 0; a < 2; ++a)
        #pragma unroll
        for (int b = 0; b < 2; ++b)
            #pragma unroll
            for (int c = 0; c < 2; ++c) {
                int f = (((z << 1) + a) * Hf + ((y << 1) + b)) * Wf + ((x << 1) + c);
                v = fmaxf(v, fine[f]);
            }
    coarse[i] = v;
}

// ---------------------------------------------------------------------------
// Host orchestration: whole U-Net as a launch sequence on `stream`.
// Param leaf order follows setup_inputs() insertion order.
// ---------------------------------------------------------------------------
extern "C" void kernel_launch(void* const* d_in, const int* in_sizes, int n_in,
                              void* d_out, int out_size, void* d_ws, size_t ws_size,
                              hipStream_t stream) {
    (void)in_sizes; (void)n_in; (void)out_size; (void)ws_size;
    const float* x = (const float*)d_in[0];
    const unsigned char* maskraw = (const unsigned char*)d_in[1];
    auto F = [&](int i) { return (const float*)d_in[i]; };

    const int D0 = 16, H0 = 96, W0 = 96, P0 = D0 * H0 * W0;
    const int D1 = 8,  H1 = 48, W1 = 48, P1 = D1 * H1 * W1;
    const int D2 = 4,  H2 = 24, W2 = 24, P2 = D2 * H2 * W2;
    const int D3 = 2,  H3 = 12, W3 = 12, P3 = D3 * H3 * W3;

    float* ws = (float*)d_ws;
    size_t cur = 0;
    auto alloc = [&](size_t n) { float* p = ws + cur; cur += n; return p; };

    float* m0 = alloc(P0);
    float* m1 = alloc(P1);
    float* m2 = alloc(P2);
    float* m3 = alloc(P3);
    float* t0a = alloc((size_t)32 * P0);
    float* t0b = alloc((size_t)32 * P0);
    float* e0  = alloc((size_t)32 * P0);
    float* u0  = alloc((size_t)32 * P0);
    float* t1a = alloc((size_t)64 * P1);
    float* t1b = alloc((size_t)64 * P1);
    float* e1  = alloc((size_t)64 * P1);
    float* u1  = alloc((size_t)64 * P1);
    float* t2a = alloc((size_t)128 * P2);
    float* t2b = alloc((size_t)128 * P2);
    float* e2  = alloc((size_t)128 * P2);
    float* u2  = alloc((size_t)128 * P2);
    float* t3a = alloc((size_t)256 * P3);
    float* t3b = alloc((size_t)256 * P3);

    mask_init_kernel<<<(P0 + 255) / 256, 256, 0, stream>>>(maskraw, m0, P0);
    mask_pool_kernel<<<(P1 + 255) / 256, 256, 0, stream>>>(m0, m1, D1, H1, W1);
    mask_pool_kernel<<<(P2 + 255) / 256, 256, 0, stream>>>(m1, m2, D2, H2, W2);
    mask_pool_kernel<<<(P3 + 255) / 256, 256, 0, stream>>>(m2, m3, D3, H3, W3);

    auto log2i = [](int v) { int s = 0; while ((1 << s) < v) ++s; return s; };

    auto conv = [&](const float* inA, const float* inB, int cinA, int cinTot, int taps,
                    const float* w, int Cout, const float* s, const float* b,
                    const float* msk, float* out, int D, int H, int W, int kmode) {
        int numXT = (W + 31) / 32;
        int nCi = cinTot > 32 ? cinTot / 32 : 1;
        int nG = Cout / 32;
        dim3 grid(numXT * nG, H, D);
        conv_wmma_kernel<<<grid, 64, 0, stream>>>(inA, inB, cinA, cinTot, taps,
                                                  w, s, b, msk, out, D, H, W, kmode,
                                                  log2i(nCi), log2i(nG));
    };

    // encoder
    conv(x,   nullptr, 4,  4,  27, F(2),  32, F(3),  F(4),  m0, t0a, D0, H0, W0, 0);
    conv(t0a, nullptr, 32, 32, 27, F(5),  32, F(6),  F(7),  m0, e0,  D0, H0, W0, 0);
    conv(e0,  nullptr, 32, 32, 8,  F(8),  64, F(9),  F(10), m1, t1a, D1, H1, W1, 1);
    conv(t1a, nullptr, 64, 64, 27, F(11), 64, F(12), F(13), m1, t1b, D1, H1, W1, 0);
    conv(t1b, nullptr, 64, 64, 27, F(14), 64, F(15), F(16), m1, e1,  D1, H1, W1, 0);
    conv(e1,  nullptr, 64, 64, 8,  F(17), 128, F(18), F(19), m2, t2a, D2, H2, W2, 1);
    conv(t2a, nullptr, 128, 128, 27, F(20), 128, F(21), F(22), m2, t2b, D2, H2, W2, 0);
    conv(t2b, nullptr, 128, 128, 27, F(23), 128, F(24), F(25), m2, e2,  D2, H2, W2, 0);
    conv(e2,  nullptr, 128, 128, 8,  F(26), 256, F(27), F(28), m3, t3a, D3, H3, W3, 1);
    // bottleneck
    conv(t3a, nullptr, 256, 256, 27, F(29), 256, F(30), F(31), m3, t3b, D3, H3, W3, 0);
    conv(t3b, nullptr, 256, 256, 27, F(32), 256, F(33), F(34), m3, t3a, D3, H3, W3, 0);
    // decoder
    upconv_kernel<<<((size_t)128 * P2 + 255) / 256, 256, 0, stream>>>(
        t3a, F(35), m2, u2, 256, 128, D2, H2, W2);
    conv(u2,  e2,      128, 256, 27, F(36), 128, F(37), F(38), m2, t2a, D2, H2, W2, 0);
    conv(t2a, nullptr, 128, 128, 27, F(39), 128, F(40), F(41), m2, t2b, D2, H2, W2, 0);
    upconv_kernel<<<((size_t)64 * P1 + 255) / 256, 256, 0, stream>>>(
        t2b, F(42), m1, u1, 128, 64, D1, H1, W1);
    conv(u1,  e1,      64, 128, 27, F(43), 64, F(44), F(45), m1, t1a, D1, H1, W1, 0);
    conv(t1a, nullptr, 64, 64,  27, F(46), 64, F(47), F(48), m1, t1b, D1, H1, W1, 0);
    upconv_kernel<<<((size_t)32 * P0 + 255) / 256, 256, 0, stream>>>(
        t1b, F(49), m0, u0, 64, 32, D0, H0, W0);
    conv(u0,  e0,      32, 64, 27, F(50), 32, F(51), F(52), m0, t0a, D0, H0, W0, 0);
    conv(t0a, nullptr, 32, 32, 27, F(53), 32, F(54), F(55), m0, t0b, D0, H0, W0, 0);
    // head
    pred_kernel<<<(P0 + 255) / 256, 256, 0, stream>>>(
        t0b, F(56), F(57), m0, (float*)d_out, P0);
}
